// SMVDR_70076686402471
// MI455X (gfx1250) — matrix operand order repaired and checked by reading, not compiled, not run on hardware
//
#include <hip/hip_runtime.h>
#include <hip/hip_bf16.h>

// ---------------- problem constants (match reference) ----------------
#define LNUM 4
#define BATCH 64
#define NTOK 196
#define CDIM 768
#define CNEW 960
#define HIDD 3072
#define NHEAD 12
#define HD 64
#define HDK 80
#define HWDIM 14
#define M49 49
#define BN (BATCH * NTOK)   // 12544
#define BM (BATCH * M49)    // 3136

typedef __attribute__((ext_vector_type(16))) __bf16 v16bf;
typedef __attribute__((ext_vector_type(8)))  __bf16 v8bf;
typedef __attribute__((ext_vector_type(8)))  float  v8f;
typedef __attribute__((ext_vector_type(4)))  unsigned int u32x4;
typedef __attribute__((ext_vector_type(4)))  int i32x4;
typedef __attribute__((ext_vector_type(8)))  int i32x8;

#define TILE_M 64
#define TILE_N 64
#define TILE_K 32
#define LDSTR 40   // LDS row stride in bf16 elems: 80B -> 16B-aligned frags, bank spread

// TDM availability (device pass only; host pass falls back and discards body)
#if defined(__has_builtin)
#if __has_builtin(__builtin_amdgcn_tensor_load_to_lds) && __has_builtin(__builtin_amdgcn_s_wait_tensorcnt)
#define USE_TDM 1
#else
#define USE_TDM 0
#endif
#else
#define USE_TDM 0
#endif

__device__ __forceinline__ float gelu_exact(float v) {
    return 0.5f * v * (1.0f + erff(v * 0.70710678118654752f));
}

// ---------------- fp32 (K x Nc) -> bf16 transposed (Nc x K) ----------------
// grid: (K/32, Nc/32, LNUM), block: 256
__global__ __launch_bounds__(256)
void f2bf_t_kernel(const float* __restrict__ in, __hip_bfloat16* __restrict__ out,
                   int K, int Nc) {
    __shared__ float t[32][33];
    const size_t lbase = (size_t)blockIdx.z * K * Nc;
    const int k0 = blockIdx.x * 32, n0 = blockIdx.y * 32;
    const int tx = threadIdx.x & 31, ty = threadIdx.x >> 5;  // ty: 0..7
#pragma unroll
    for (int i = ty; i < 32; i += 8)
        t[i][tx] = in[lbase + (size_t)(k0 + i) * Nc + (n0 + tx)];
    __syncthreads();
#pragma unroll
    for (int i = ty; i < 32; i += 8)
        out[lbase + (size_t)(n0 + i) * K + (k0 + tx)] = __float2bfloat16(t[tx][i]);
}

// ---------------- LayerNorm (+optional exact GELU) -> bf16 ----------------
__global__ __launch_bounds__(256)
void ln_bf16_kernel(const float* __restrict__ x, const float* __restrict__ g,
                    const float* __restrict__ bta, float eps, int cols, int gelu,
                    __hip_bfloat16* __restrict__ out) {
    __shared__ float s1[256];
    __shared__ float s2[256];
    const int row = blockIdx.x;
    const float* xr = x + (size_t)row * cols;
    float a = 0.f, b = 0.f;
    for (int c = threadIdx.x; c < cols; c += 256) { float t = xr[c]; a += t; b += t * t; }
    s1[threadIdx.x] = a; s2[threadIdx.x] = b;
    __syncthreads();
    for (int off = 128; off > 0; off >>= 1) {
        if ((int)threadIdx.x < off) {
            s1[threadIdx.x] += s1[threadIdx.x + off];
            s2[threadIdx.x] += s2[threadIdx.x + off];
        }
        __syncthreads();
    }
    const float mean = s1[0] / (float)cols;
    const float var  = s2[0] / (float)cols - mean * mean;
    const float rs   = rsqrtf(var + eps);
    __hip_bfloat16* orow = out + (size_t)row * cols;
    for (int c = threadIdx.x; c < cols; c += 256) {
        float v = (xr[c] - mean) * rs * g[c] + bta[c];
        if (gelu) v = gelu_exact(v);
        orow[c] = __float2bfloat16(v);
    }
}

// ---------------- 2x2 stride-2 depthwise conv (groups=C) ----------------
__global__ __launch_bounds__(256)
void dwconv_kernel(const __hip_bfloat16* __restrict__ y, const float* __restrict__ w,
                   const float* __restrict__ bvec, __hip_bfloat16* __restrict__ r0) {
    const int c  = blockIdx.x * 256 + threadIdx.x;
    const int bm = blockIdx.y;
    const int b  = bm / M49, m = bm % M49;
    const int oh = m / 7, ow = m % 7;
    float acc = bvec[c];
#pragma unroll
    for (int i = 0; i < 2; ++i)
#pragma unroll
        for (int j = 0; j < 2; ++j) {
            const int tok = (2 * oh + i) * HWDIM + (2 * ow + j);
            acc += __bfloat162float(y[((size_t)(b * NTOK + tok)) * CDIM + c]) *
                   w[c * 4 + i * 2 + j];
        }
    r0[(size_t)bm * CDIM + c] = __float2bfloat16(acc);
}

// ---------------- WMMA bf16 GEMM ----------------
// C(Mr x Nc) = A(Mr x K, row-major) * Bt(Nc x K, row-major = pre-transposed W)
// FLAGS: 1=bias  2=gelu  4=residual-add(fp32)  8=write fp32  16=write bf16
template <int FLAGS>
__global__ __launch_bounds__(128)
void gemm_bf16_wmma(const __hip_bfloat16* __restrict__ A,
                    const __hip_bfloat16* __restrict__ Bt,
                    const float* __restrict__ bias,
                    const float* __restrict__ resid,
                    float* __restrict__ outF,
                    __hip_bfloat16* __restrict__ outB,
                    int Mr, int Nc, int K) {
    __shared__ __align__(16) __hip_bfloat16 As[TILE_M * LDSTR];  // 5 KB
    __shared__ __align__(16) __hip_bfloat16 Bs[TILE_N * LDSTR];  // 5 KB

    const int tid   = threadIdx.x;
    const int wave  = tid >> 5;
    const int lane  = tid & 31;
    const int tileM = blockIdx.y * TILE_M;
    const int tileN = blockIdx.x * TILE_N;
    const int wm = (wave >> 1) * 32;
    const int wn = (wave & 1) * 32;

    v8f acc[2][2] = {};

    const int srow = tid >> 1;   // 0..63
    const int sseg = tid & 1;    // 0..1

#if USE_TDM
    // D# group1: constant tensor/tile descriptor for the B tile
    // data_size=2B, pad_enable, pad_interval=16 dwords (one 64B row), pad_amount=4 dwords
    i32x8 g1;
    g1[0] = (int)(0x00010000u | (1u << 20) | (3u << 22) | (3u << 25));
    g1[1] = (int)((unsigned)(K & 0xFFFF) << 16);                          // tensor_dim0 lo
    g1[2] = (int)(((unsigned)K >> 16) | ((unsigned)(Nc & 0xFFFF) << 16)); // dim0 hi | dim1 lo
    g1[3] = (int)(((unsigned)Nc >> 16) | ((unsigned)TILE_K << 16));       // dim1 hi | tile_dim0
    g1[4] = TILE_N;                                                       // tile_dim1
    g1[5] = K;                                                            // tensor_dim0_stride
    g1[6] = 0;
    g1[7] = 0;
    const i32x4 z4 = {0, 0, 0, 0};
    const unsigned ldsB = (unsigned)(size_t)(&Bs[0]);
#if __clang_major__ >= 23
    const i32x8 z8 = {0, 0, 0, 0, 0, 0, 0, 0};
#endif
#endif

    for (int k0 = 0; k0 < K; k0 += TILE_K) {
        // ---- stage A tile (64 x 32 bf16): two 16B vector copies per thread
        {
            const uint4* src =
                (const uint4*)(A + (size_t)(tileM + srow) * K + k0) + sseg * 2;
            uint4* dst = (uint4*)(As + srow * LDSTR) + sseg * 2;
            dst[0] = src[0];
            dst[1] = src[1];
            if (k0 + TILE_K < K)
                __builtin_prefetch(A + (size_t)(tileM + srow) * K + k0 + TILE_K, 0, 1);
        }
#if USE_TDM
        // ---- stage B tile via Tensor Data Mover (one issue per block)
        if (wave == 0) {
            const unsigned long long ga =
                (unsigned long long)(size_t)(Bt + (size_t)tileN * K + k0);
            u32x4 g0;
            g0[0] = 1u;                                        // count=1, user mode
            g0[1] = ldsB;                                      // lds_addr
            g0[2] = (unsigned)(ga & 0xFFFFFFFFu);              // global_addr lo
            g0[3] = (unsigned)((ga >> 32) & 0x1FFFFFFu) | 0x80000000u;  // hi | type=2
#if __clang_major__ >= 23
            __builtin_amdgcn_tensor_load_to_lds(g0, g1, z4, z4, z8, 0);
#else
            __builtin_amdgcn_tensor_load_to_lds(g0, g1, z4, z4, 0);
#endif
            __builtin_amdgcn_s_wait_tensorcnt((short)0);
        }
#else
        // ---- fallback: manual B tile staging (Bt is row-major Nc x K)
        {
            const uint4* src =
                (const uint4*)(Bt + (size_t)(tileN + srow) * K + k0) + sseg * 2;
            uint4* dst = (uint4*)(Bs + srow * LDSTR) + sseg * 2;
            dst[0] = src[0];
            dst[1] = src[1];
        }
#endif
        __syncthreads();

        // ---- fragments per ISA 16-bit WMMA layouts
        v16bf afr[2], bfr[2];
        {
            const int ar   = wm + (lane & 15);
            const int koff = (lane >> 4) * 8;  // A: lanes 0-15 K0-7/16-23, 16-31 K8-15/24-31
#pragma unroll
            for (int t = 0; t < 2; ++t) {
                const v8bf lo = *(const v8bf*)&As[(ar + t * 16) * LDSTR + koff];
                const v8bf hi = *(const v8bf*)&As[(ar + t * 16) * LDSTR + koff + 16];
                v16bf a;
#pragma unroll
                for (int i = 0; i < 8; ++i) { a[i] = lo[i]; a[i + 8] = hi[i]; }
                afr[t] = a;
            }
            const int bc = wn + (lane & 15);
            const int kb = (lane >> 4) * 8;    // B: lanes 0-15 K0-7/16-23, 16-31 K8-15/24-31
#pragma unroll
            for (int t = 0; t < 2; ++t) {
                const v8bf lo = *(const v8bf*)&Bs[(bc + t * 16) * LDSTR + kb];
                const v8bf hi = *(const v8bf*)&Bs[(bc + t * 16) * LDSTR + kb + 16];
                v16bf b;
#pragma unroll
                for (int i = 0; i < 8; ++i) { b[i] = lo[i]; b[i + 8] = hi[i]; }
                bfr[t] = b;
            }
        }
#pragma unroll
        for (int i = 0; i < 2; ++i)
#pragma unroll
            for (int j = 0; j < 2; ++j)
                acc[i][j] = __builtin_amdgcn_wmma_f32_16x16x32_bf16(
                    false, afr[i], false, bfr[j], (short)0, acc[i][j], false, false);
        __syncthreads();
    }

    // ---- epilogue: C/D layout -> row = r + 8*(lane>=16), col = lane&15
    const int rbase = (lane >> 4) * 8;
    const int cidx  = lane & 15;
#pragma unroll
    for (int i = 0; i < 2; ++i) {
#pragma unroll
        for (int j = 0; j < 2; ++j) {
            const int gN = tileN + wn + j * 16 + cidx;
#pragma unroll
            for (int r = 0; r < 8; ++r) {
                const int gM = tileM + wm + i * 16 + rbase + r;
                float v = acc[i][j][r];
                if (FLAGS & 1) v += bias[gN];
                if (FLAGS & 2) v = gelu_exact(v);
                const size_t idx = (size_t)gM * Nc + gN;
                if (FLAGS & 4) v += resid[idx];
                if (FLAGS & 8)  outF[idx] = v;
                if (FLAGS & 16) outB[idx] = __float2bfloat16(v);
            }
        }
    }
}

// ---------------- attention: per (b,h), 196 queries x 49 kv ----------------
__global__ __launch_bounds__(256)
void attn_kernel(const float* __restrict__ qbuf, const float* __restrict__ kbuf,
                 const float* __restrict__ vbuf, __hip_bfloat16* __restrict__ obuf) {
    const int bh = blockIdx.x;
    const int b = bh / NHEAD, h = bh % NHEAD;
    __shared__ float ks[M49 * HDK];
    __shared__ float vs[M49 * HD];
    for (int i = threadIdx.x; i < M49 * HDK; i += 256)
        ks[i] = kbuf[(size_t)(b * M49 + i / HDK) * CNEW + h * HDK + (i % HDK)];
    for (int i = threadIdx.x; i < M49 * HD; i += 256)
        vs[i] = vbuf[(size_t)(b * M49 + i / HD) * CDIM + h * HD + (i % HD)];
    __syncthreads();
    const int n = threadIdx.x;
    if (n < NTOK) {
        const float* qp = qbuf + (size_t)(b * NTOK + n) * CNEW + h * HDK;
        float s[M49];
        float mx = -3.0e30f;
        const float scale = 0.111803398874989485f;  // 1/sqrt(80)
        for (int m = 0; m < M49; ++m) {
            float acc = 0.f;
            for (int d = 0; d < HDK; ++d) acc += qp[d] * ks[m * HDK + d];
            acc *= scale;
            s[m] = acc;
            mx = fmaxf(mx, acc);
        }
        float sum = 0.f;
        for (int m = 0; m < M49; ++m) { s[m] = expf(s[m] - mx); sum += s[m]; }
        const float inv = 1.f / sum;
        __hip_bfloat16* op = obuf + (size_t)(b * NTOK + n) * CDIM + h * HD;
        for (int d = 0; d < HD; ++d) {
            float acc = 0.f;
            for (int m = 0; m < M49; ++m) acc += s[m] * vs[m * HD + d];
            op[d] = __float2bfloat16(acc * inv);
        }
    }
}

// ---------------- host orchestration ----------------
static void launch_gemm(int flags, hipStream_t stream,
                        const __hip_bfloat16* A, const __hip_bfloat16* Bt,
                        const float* bias, const float* resid,
                        float* outF, __hip_bfloat16* outB, int Mr, int Nc, int K) {
    dim3 grid(Nc / TILE_N, Mr / TILE_M), block(128);
    switch (flags) {
        case 8:
            gemm_bf16_wmma<8><<<grid, block, 0, stream>>>(A, Bt, bias, resid, outF, outB, Mr, Nc, K);
            break;
        case 9:
            gemm_bf16_wmma<9><<<grid, block, 0, stream>>>(A, Bt, bias, resid, outF, outB, Mr, Nc, K);
            break;
        case 13:
            gemm_bf16_wmma<13><<<grid, block, 0, stream>>>(A, Bt, bias, resid, outF, outB, Mr, Nc, K);
            break;
        case 19:
            gemm_bf16_wmma<19><<<grid, block, 0, stream>>>(A, Bt, bias, resid, outF, outB, Mr, Nc, K);
            break;
        default: break;
    }
}

extern "C" void kernel_launch(void* const* d_in, const int* in_sizes, int n_in,
                              void* d_out, int out_size, void* d_ws, size_t ws_size,
                              hipStream_t stream) {
    (void)in_sizes; (void)n_in; (void)out_size; (void)ws_size;
    typedef __hip_bfloat16 bf;

    const float* in_x     = (const float*)d_in[0];
    const float* in_qw    = (const float*)d_in[1];
    const float* in_dww   = (const float*)d_in[2];
    const float* in_dwb   = (const float*)d_in[3];
    const float* in_pww   = (const float*)d_in[4];
    const float* in_pwb   = (const float*)d_in[5];
    const float* in_lnrg  = (const float*)d_in[6];
    const float* in_lnrb  = (const float*)d_in[7];
    const float* in_kw    = (const float*)d_in[8];
    const float* in_vw    = (const float*)d_in[9];
    const float* in_pjw   = (const float*)d_in[10];
    const float* in_pjb   = (const float*)d_in[11];
    const float* in_ln1g  = (const float*)d_in[12];
    const float* in_ln1b  = (const float*)d_in[13];
    const float* in_ln2g  = (const float*)d_in[14];
    const float* in_ln2b  = (const float*)d_in[15];
    const float* in_fc1w  = (const float*)d_in[16];
    const float* in_fc1b  = (const float*)d_in[17];
    const float* in_fc2w  = (const float*)d_in[18];
    const float* in_fc2b  = (const float*)d_in[19];

    // ---- carve workspace
    char* ws = (char*)d_ws;
    auto carve = [&](size_t bytes) -> char* {
        char* p = ws;
        ws += (bytes + 255) & ~(size_t)255;
        return p;
    };
    float* xbuf   = (float*)carve((size_t)BN * CDIM * 4);
    bf*    y_bf   = (bf*)carve((size_t)BN * CDIM * 2);
    float* q_f    = (float*)carve((size_t)BN * CNEW * 4);
    bf*    r0_bf  = (bf*)carve((size_t)BM * CDIM * 2);
    float* rpre_f = (float*)carve((size_t)BM * CNEW * 4);
    bf*    r_bf   = (bf*)carve((size_t)BM * CNEW * 2);
    float* k_f    = (float*)carve((size_t)BM * CNEW * 4);
    float* v_f    = (float*)carve((size_t)BM * CDIM * 4);
    bf*    o_bf   = (bf*)carve((size_t)BN * CDIM * 2);
    bf*    z_bf   = (bf*)carve((size_t)BN * CDIM * 2);
    bf*    h_bf   = (bf*)carve((size_t)BN * HIDD * 2);
    bf* qw_bf  = (bf*)carve((size_t)LNUM * CDIM * CNEW * 2);
    bf* pww_bf = (bf*)carve((size_t)LNUM * CDIM * CNEW * 2);
    bf* kw_bf  = (bf*)carve((size_t)LNUM * CNEW * CNEW * 2);
    bf* vw_bf  = (bf*)carve((size_t)LNUM * CNEW * CDIM * 2);
    bf* pjw_bf = (bf*)carve((size_t)LNUM * CDIM * CDIM * 2);
    bf* f1w_bf = (bf*)carve((size_t)LNUM * CDIM * HIDD * 2);
    bf* f2w_bf = (bf*)carve((size_t)LNUM * HIDD * CDIM * 2);

    // ---- convert + transpose all weights to bf16 (Nc x K per layer)
    auto convt = [&](const float* src, bf* dst, int K, int Nc) {
        dim3 grid(K / 32, Nc / 32, LNUM), block(256);
        f2bf_t_kernel<<<grid, block, 0, stream>>>(src, dst, K, Nc);
    };
    convt(in_qw,   qw_bf,  CDIM, CNEW);
    convt(in_pww,  pww_bf, CDIM, CNEW);
    convt(in_kw,   kw_bf,  CNEW, CNEW);
    convt(in_vw,   vw_bf,  CNEW, CDIM);
    convt(in_pjw,  pjw_bf, CDIM, CDIM);
    convt(in_fc1w, f1w_bf, CDIM, HIDD);
    convt(in_fc2w, f2w_bf, HIDD, CDIM);

    // residual stream lives in xbuf (never mutate d_in)
    hipMemcpyAsync(xbuf, in_x, (size_t)BN * CDIM * 4, hipMemcpyDeviceToDevice, stream);

    for (int l = 0; l < LNUM; ++l) {
        // ----- attention branch -----
        ln_bf16_kernel<<<dim3(BN), dim3(256), 0, stream>>>(
            xbuf, in_ln1g + (size_t)l * CDIM, in_ln1b + (size_t)l * CDIM,
            1e-6f, CDIM, 0, y_bf);
        launch_gemm(8, stream, y_bf, qw_bf + (size_t)l * CDIM * CNEW,
                    nullptr, nullptr, q_f, nullptr, BN, CNEW, CDIM);
        dwconv_kernel<<<dim3(CDIM / 256, BM), dim3(256), 0, stream>>>(
            y_bf, in_dww + (size_t)l * CDIM * 4, in_dwb + (size_t)l * CDIM, r0_bf);
        launch_gemm(9, stream, r0_bf, pww_bf + (size_t)l * CDIM * CNEW,
                    in_pwb + (size_t)l * CNEW, nullptr, rpre_f, nullptr, BM, CNEW, CDIM);
        ln_bf16_kernel<<<dim3(BM), dim3(256), 0, stream>>>(
            rpre_f, in_lnrg + (size_t)l * CNEW, in_lnrb + (size_t)l * CNEW,
            1e-5f, CNEW, 1, r_bf);
        launch_gemm(8, stream, r_bf, kw_bf + (size_t)l * CNEW * CNEW,
                    nullptr, nullptr, k_f, nullptr, BM, CNEW, CNEW);
        launch_gemm(8, stream, r_bf, vw_bf + (size_t)l * CNEW * CDIM,
                    nullptr, nullptr, v_f, nullptr, BM, CDIM, CNEW);
        attn_kernel<<<dim3(BATCH * NHEAD), dim3(256), 0, stream>>>(q_f, k_f, v_f, o_bf);
        launch_gemm(13, stream, o_bf, pjw_bf + (size_t)l * CDIM * CDIM,
                    in_pjb + (size_t)l * CDIM, xbuf, xbuf, nullptr, BN, CDIM, CDIM);

        // ----- MLP branch -----
        ln_bf16_kernel<<<dim3(BN), dim3(256), 0, stream>>>(
            xbuf, in_ln2g + (size_t)l * CDIM, in_ln2b + (size_t)l * CDIM,
            1e-6f, CDIM, 0, z_bf);
        launch_gemm(19, stream, z_bf, f1w_bf + (size_t)l * CDIM * HIDD,
                    in_fc1b + (size_t)l * HIDD, nullptr, nullptr, h_bf, BN, HIDD, CDIM);
        launch_gemm(13, stream, h_bf, f2w_bf + (size_t)l * HIDD * CDIM,
                    in_fc2b + (size_t)l * CDIM, xbuf, xbuf, nullptr, BN, CDIM, HIDD);
    }

    hipMemcpyAsync(d_out, xbuf, (size_t)BN * CDIM * 4, hipMemcpyDeviceToDevice, stream);
}
